// BehavioralRotaryAttentionV25_47648367182296
// MI455X (gfx1250) — compile-verified, matrix operand order checked by reading, not run on hardware
//
#include <hip/hip_runtime.h>
#include <hip/hip_bf16.h>
#include <math.h>

// ---------------------------------------------------------------------------
// MI455X / gfx1250 rotary-attention block.
// - All GEMMs + attention matmuls: v_wmma_f32_16x16x32_bf16 (f32 accumulate).
// - Attention stages K/V chunks into LDS via GLOBAL_LOAD_ASYNC_TO_LDS_B128
//   (double-buffered, ASYNCcnt-tracked) when the toolchain exposes the
//   builtin; falls back to direct global fragment loads otherwise.
// ---------------------------------------------------------------------------

typedef __attribute__((ext_vector_type(16))) __bf16    v16bf;
typedef __attribute__((ext_vector_type(8)))  float     v8f;
typedef __attribute__((ext_vector_type(4)))  float     v4f;
typedef __attribute__((ext_vector_type(2)))  unsigned  v2u;
typedef __attribute__((ext_vector_type(4)))  int       v4i;

#if defined(__HIP_DEVICE_COMPILE__) && __has_builtin(__builtin_amdgcn_global_load_async_to_lds_b128)
#define USE_ASYNC_LDS 1
#else
#define USE_ASYNC_LDS 0
#endif

#if USE_ASYNC_LDS
typedef __attribute__((address_space(1))) v4i as1_v4i;
typedef __attribute__((address_space(3))) v4i as3_v4i;
__device__ __forceinline__ void async_load_b128(const void* g, void* l) {
    // (global src v4i*, LDS dst v4i*, imm offset, imm cpol)
    __builtin_amdgcn_global_load_async_to_lds_b128(
        (as1_v4i*)(void*)g, (as3_v4i*)l, 0, 0);
}
#endif

union FragBF {
    v16bf v;
    unsigned short u[16];
    v4f f4[2];
};

__device__ __forceinline__ unsigned short f2bf(float x) {
    unsigned u = __float_as_uint(x);
    u = u + 0x7FFFu + ((u >> 16) & 1u);      // round-to-nearest-even
    return (unsigned short)(u >> 16);
}

__device__ __forceinline__ v8f v8f_zero() {
    v8f z;
#pragma unroll
    for (int i = 0; i < 8; ++i) z[i] = 0.0f;
    return z;
}

// Load one 16x32 bf16 WMMA fragment from a row-major [rows x ld] bf16 matrix.
// Works for A fragments (lane = M row) and B fragments (lane = N col):
//   lane<16 : K = k0+0..7  and k0+16..23
//   lane>=16: K = k0+8..15 and k0+24..31
// Also used against LDS tiles (addrspace inferred after inlining -> ds_load).
__device__ __forceinline__ v16bf load_frag(const unsigned short* __restrict__ p,
                                           int row0, int ld, int k0, int lane) {
    const int r   = row0 + (lane & 15);
    const int sel = (lane >> 4) & 1;
    const unsigned short* base = p + (size_t)r * ld + k0 + sel * 8;
    FragBF f;
    f.f4[0] = *(const v4f*)(base);
    f.f4[1] = *(const v4f*)(base + 16);
    return f.v;
}

__device__ __forceinline__ v8f wmma_bf16(v16bf a, v16bf b, v8f c) {
    return __builtin_amdgcn_wmma_f32_16x16x32_bf16(
        /*neg_a=*/false, a, /*neg_b=*/false, b,
        /*c_mod=*/(short)0, c, /*reuse_a=*/false, /*reuse_b=*/false);
}

// ---------------------------------------------------------------------------
// f32 -> bf16 conversion (vectorized 4-wide)
// ---------------------------------------------------------------------------
__global__ void convert_f32_bf16_k(const v4f* __restrict__ in,
                                   v2u* __restrict__ out, int n4) {
    int i = blockIdx.x * blockDim.x + threadIdx.x;
    if (i >= n4) return;
    v4f f = in[i];
    v2u o;
    o.x = (unsigned)f2bf(f[0]) | ((unsigned)f2bf(f[1]) << 16);
    o.y = (unsigned)f2bf(f[2]) | ((unsigned)f2bf(f[3]) << 16);
    out[i] = o;
}

// ---------------------------------------------------------------------------
// GEMM: C[N x M] = A[N x K](bf16) @ Bw[M x K](bf16)^T + bias (+ residual)
// 128 threads = 4 waves, 64x64 tile per wave -> 128x128 per block.
// Direct global fragment loads: weights (2 MB) and activation tiles are
// L2-resident (192 MB L2); compiler software-pipelines load clauses vs WMMA.
// ---------------------------------------------------------------------------
__global__ __launch_bounds__(128)
void gemm_bf16_k(const unsigned short* __restrict__ A,
                 const unsigned short* __restrict__ Bw,
                 const float* __restrict__ bias,
                 const float* __restrict__ residual,
                 float* __restrict__ C,
                 int N, int M, int K) {
    const int lane = threadIdx.x & 31;
    const int wave = threadIdx.x >> 5;
    const int row0 = blockIdx.y * 128 + (wave >> 1) * 64;
    const int col0 = blockIdx.x * 128 + (wave & 1) * 64;

    v8f acc[4][4];
#pragma unroll
    for (int i = 0; i < 4; ++i)
#pragma unroll
        for (int t = 0; t < 4; ++t) acc[i][t] = v8f_zero();

    for (int k0 = 0; k0 < K; k0 += 32) {
        v16bf a[4], b[4];
#pragma unroll
        for (int i = 0; i < 4; ++i) a[i] = load_frag(A,  row0 + 16 * i, K, k0, lane);
#pragma unroll
        for (int t = 0; t < 4; ++t) b[t] = load_frag(Bw, col0 + 16 * t, K, k0, lane);
#pragma unroll
        for (int i = 0; i < 4; ++i)
#pragma unroll
            for (int t = 0; t < 4; ++t)
                acc[i][t] = wmma_bf16(a[i], b[t], acc[i][t]);
    }

    const int cl   = lane & 15;
    const int rofs = (lane & 16) ? 8 : 0;
#pragma unroll
    for (int t = 0; t < 4; ++t) {
        const int col = col0 + 16 * t + cl;
        const float bv = bias[col];
#pragma unroll
        for (int i = 0; i < 4; ++i) {
#pragma unroll
            for (int j = 0; j < 8; ++j) {
                const int row = row0 + 16 * i + rofs + j;
                const size_t idx = (size_t)row * M + col;
                float val = acc[i][t][j] + bv;
                if (residual) val += residual[idx];
                C[idx] = val;
            }
        }
    }
}

// ---------------------------------------------------------------------------
// Rotary (per-token scalar phase) + repack to head-major bf16.
// q,k,v: f32 [B, L, H*64].  Outputs: qr,kr bf16 [B,H,L,64]; vt bf16 [B,H,64,L].
// ---------------------------------------------------------------------------
__global__ void rotary_k(const float* __restrict__ q, const float* __restrict__ k,
                         const float* __restrict__ v, const float* __restrict__ phi,
                         unsigned short* __restrict__ qr, unsigned short* __restrict__ kr,
                         unsigned short* __restrict__ vt, int B, int H, int L) {
    int idx = blockIdx.x * blockDim.x + threadIdx.x;
    int total = B * H * L * 32;
    if (idx >= total) return;
    int e  = idx & 31;
    int l  = (idx >> 5) % L;
    int bh = idx / (32 * L);
    int h = bh % H, b = bh / H;

    float phase = phi[(size_t)bh * L + l];
    float sp, cp;
    __sincosf(phase, &sp, &cp);

    size_t src = ((size_t)(b * L + l)) * (H * 64) + h * 64 + e;
    float q1 = q[src], q2 = q[src + 32];
    float k1 = k[src], k2 = k[src + 32];
    float v1 = v[src], v2 = v[src + 32];

    // out = x*cos + rotate_half(x)*sin ; rotate_half([x1,x2]) = [-x2, x1]
    size_t dst = ((size_t)bh * L + l) * 64 + e;
    qr[dst]      = f2bf(q1 * cp - q2 * sp);
    qr[dst + 32] = f2bf(q2 * cp + q1 * sp);
    kr[dst]      = f2bf(k1 * cp - k2 * sp);
    kr[dst + 32] = f2bf(k2 * cp + k1 * sp);

    size_t vtd = ((size_t)bh * 64 + e) * L + l;
    vt[vtd]                  = f2bf(v1);
    vt[vtd + (size_t)32 * L] = f2bf(v2);
}

// ---------------------------------------------------------------------------
// Flash-style attention, one wave per 16-query tile, 4 waves per block.
// S^T = K_chunk(32x64) @ Q^T(64x16): each lane holds all scores of one query,
// softmax needs a single shfl_xor(16), and exp(S) lands directly in the
// A-fragment layout for P @ V (no LDS transposes).
// K/V chunks are shared by the 4 waves -> staged once per block into LDS with
// double-buffered async copies (ASYNCcnt) when available.
// ---------------------------------------------------------------------------
__global__ __launch_bounds__(128)
void attention_k(const unsigned short* __restrict__ Q,   // [B,H,L,64] bf16
                 const unsigned short* __restrict__ Kr,  // [B,H,L,64] bf16
                 const unsigned short* __restrict__ VT,  // [B,H,64,L] bf16
                 const float* __restrict__ mask,         // [B,1,1,L]
                 unsigned short* __restrict__ ctx,       // [B,L,H*64] bf16
                 int B, int H, int L) {
    const int lane = threadIdx.x & 31;
    const int wave = threadIdx.x >> 5;
    const int bh   = blockIdx.y;
    const int b = bh / H, h = bh % H;
    const int q0 = (blockIdx.x * 4 + wave) * 16;

    const unsigned short* Qp = Q  + (size_t)bh * L * 64;
    const unsigned short* Kp = Kr + (size_t)bh * L * 64;
    const unsigned short* Vp = VT + (size_t)bh * 64 * L;
    const float* mp = mask + (size_t)b * L;

    const v16bf bq0 = load_frag(Qp, q0, 64, 0,  lane);   // Q^T B-frags (lane = query)
    const v16bf bq1 = load_frag(Qp, q0, 64, 32, lane);

    v8f O[4];
#pragma unroll
    for (int t = 0; t < 4; ++t) O[t] = v8f_zero();

    float mrun = -INFINITY, lrun = 0.0f;
    const float scale = 0.125f;                 // 1/sqrt(64)
    const int sel8 = (lane & 16) ? 8 : 0;
    const int nch = L / 32;

#if USE_ASYNC_LDS
    __shared__ unsigned short ksh[2][32 * 64];  // K chunk, row-major ld=64
    __shared__ unsigned short vsh[2][64 * 32];  // V^T chunk, row-major ld=32
    const int tid = threadIdx.x;
    {   // prefetch chunk 0
#pragma unroll
        for (int p = 0; p < 2; ++p) {           // K: contiguous 4 KB
            const int i = (tid + p * 128) * 8;
            async_load_b128(Kp + i, &ksh[0][i]);
        }
#pragma unroll
        for (int p = 0; p < 2; ++p) {           // V^T: 64 rows x 64 B
            const int c = tid + p * 128;
            const int row = c >> 2, part = c & 3;
            async_load_b128(Vp + (size_t)row * L + part * 8, &vsh[0][c * 8]);
        }
    }
#endif

    for (int ic = 0; ic < nch; ++ic) {
        const int kc = ic * 32;
        v8f s0 = v8f_zero(), s1 = v8f_zero();

#if USE_ASYNC_LDS
        const int cur = ic & 1, nxt = cur ^ 1;
        if (ic + 1 < nch) {                     // prefetch next chunk
            const int kn = kc + 32;
#pragma unroll
            for (int p = 0; p < 2; ++p) {
                const int i = (tid + p * 128) * 8;
                async_load_b128(Kp + kn * 64 + i, &ksh[nxt][i]);
            }
#pragma unroll
            for (int p = 0; p < 2; ++p) {
                const int c = tid + p * 128;
                const int row = c >> 2, part = c & 3;
                async_load_b128(Vp + (size_t)row * L + kn + part * 8, &vsh[nxt][c * 8]);
            }
            asm volatile("s_wait_asynccnt 0x4" ::: "memory"); // current chunk done
        } else {
            asm volatile("s_wait_asynccnt 0x0" ::: "memory");
        }
        __syncthreads();                        // all waves' copies visible

        const unsigned short* Kc = &ksh[cur][0];
        const unsigned short* Vc = &vsh[cur][0];
        s0 = wmma_bf16(load_frag(Kc, 0,  64, 0,  lane), bq0, s0);
        s0 = wmma_bf16(load_frag(Kc, 0,  64, 32, lane), bq1, s0);
        s1 = wmma_bf16(load_frag(Kc, 16, 64, 0,  lane), bq0, s1);
        s1 = wmma_bf16(load_frag(Kc, 16, 64, 32, lane), bq1, s1);
#else
        s0 = wmma_bf16(load_frag(Kp, kc,      64, 0,  lane), bq0, s0);
        s0 = wmma_bf16(load_frag(Kp, kc,      64, 32, lane), bq1, s0);
        s1 = wmma_bf16(load_frag(Kp, kc + 16, 64, 0,  lane), bq0, s1);
        s1 = wmma_bf16(load_frag(Kp, kc + 16, 64, 32, lane), bq1, s1);
#endif

        // per-lane key indices: frag0 -> kc+sel8+j, frag1 -> kc+16+sel8+j
        v4f m0a = *(const v4f*)(mp + kc + sel8);
        v4f m0b = *(const v4f*)(mp + kc + sel8 + 4);
        v4f m1a = *(const v4f*)(mp + kc + 16 + sel8);
        v4f m1b = *(const v4f*)(mp + kc + 16 + sel8 + 4);

        float sv0[8], sv1[8];
#pragma unroll
        for (int j = 0; j < 4; ++j) {
            sv0[j]     = s0[j]     * scale + m0a[j];
            sv0[j + 4] = s0[j + 4] * scale + m0b[j];
            sv1[j]     = s1[j]     * scale + m1a[j];
            sv1[j + 4] = s1[j + 4] * scale + m1b[j];
        }

        float mx = sv0[0];
#pragma unroll
        for (int j = 1; j < 8; ++j) mx = fmaxf(mx, sv0[j]);
#pragma unroll
        for (int j = 0; j < 8; ++j) mx = fmaxf(mx, sv1[j]);
        mx = fmaxf(mx, __shfl_xor(mx, 16, 32));        // other 16 keys of this query

        const float mnew = fmaxf(mrun, mx);
        const float corr = __expf(mrun - mnew);

        FragBF pa;
        float rsum = 0.0f;
#pragma unroll
        for (int j = 0; j < 8; ++j) {
            float p = __expf(sv0[j] - mnew);
            rsum += p;
            pa.u[j] = f2bf(p);
        }
#pragma unroll
        for (int j = 0; j < 8; ++j) {
            float p = __expf(sv1[j] - mnew);
            rsum += p;
            pa.u[8 + j] = f2bf(p);
        }
        rsum += __shfl_xor(rsum, 16, 32);
        lrun = lrun * corr + rsum;
        mrun = mnew;

        // rescale O rows: O-frag row j lives in lane-half (lane>=16 -> row 8+j)
#pragma unroll
        for (int j = 0; j < 8; ++j) {
            float clo = __shfl(corr, j, 32);
            float chi = __shfl(corr, 8 + j, 32);
            float cj = (lane & 16) ? chi : clo;
            O[0][j] *= cj; O[1][j] *= cj; O[2][j] *= cj; O[3][j] *= cj;
        }

        // O(16x64) += P(16x32) @ V(32x64)
#pragma unroll
        for (int t = 0; t < 4; ++t) {
#if USE_ASYNC_LDS
            v16bf vb = load_frag(Vc, 16 * t, 32, 0, lane);
#else
            v16bf vb = load_frag(Vp, 16 * t, L, kc, lane);
#endif
            O[t] = wmma_bf16(pa.v, vb, O[t]);
        }

#if USE_ASYNC_LDS
        __syncthreads();   // done reading buf[cur]; iter ic+1 overwrites it
#endif
    }

    // normalize and store to context [B, L, H*64] as bf16
    const float rl = 1.0f / lrun;  // stats for query q0 + (lane&15)
    unsigned short* cp = ctx + (size_t)b * L * (H * 64) + h * 64;
    const int cl = lane & 15;
#pragma unroll
    for (int j = 0; j < 8; ++j) {
        float rlo = __shfl(rl, j, 32);
        float rhi = __shfl(rl, 8 + j, 32);
        float rj = (lane & 16) ? rhi : rlo;
        const int row = q0 + ((lane & 16) ? 8 : 0) + j;
        unsigned short* cr = cp + (size_t)row * (H * 64);
#pragma unroll
        for (int t = 0; t < 4; ++t)
            cr[16 * t + cl] = f2bf(O[t][j] * rj);
    }
}

// ---------------------------------------------------------------------------
// Row LayerNorm over D=1024, one 256-thread block per row.
// ---------------------------------------------------------------------------
__global__ __launch_bounds__(256)
void layernorm_k(const float* __restrict__ x, const float* __restrict__ gamma,
                 const float* __restrict__ beta, float* __restrict__ out, int D) {
    __shared__ float reds[8], reds2[8];
    const int row = blockIdx.x;
    const float* xr = x + (size_t)row * D;

    float s = 0.f, s2 = 0.f;
    for (int i = threadIdx.x; i < D; i += blockDim.x) {
        float v = xr[i];
        s += v; s2 += v * v;
    }
#pragma unroll
    for (int off = 16; off > 0; off >>= 1) {
        s  += __shfl_xor(s,  off, 32);
        s2 += __shfl_xor(s2, off, 32);
    }
    const int wv = threadIdx.x >> 5, ln = threadIdx.x & 31;
    if (ln == 0) { reds[wv] = s; reds2[wv] = s2; }
    __syncthreads();
    s = 0.f; s2 = 0.f;
#pragma unroll
    for (int w = 0; w < 8; ++w) { s += reds[w]; s2 += reds2[w]; }

    const float mu   = s / D;
    const float var  = s2 / D - mu * mu;
    const float rstd = rsqrtf(var + 1e-12f);
    for (int i = threadIdx.x; i < D; i += blockDim.x) {
        float v = (xr[i] - mu) * rstd;
        out[(size_t)row * D + i] = v * gamma[i] + beta[i];
    }
}

// ---------------------------------------------------------------------------
extern "C" void kernel_launch(void* const* d_in, const int* in_sizes, int n_in,
                              void* d_out, int out_size, void* d_ws, size_t ws_size,
                              hipStream_t stream) {
    const int B = 2, L = 2048, D = 1024, H = 16;
    const int N = B * L;                     // 4096 rows

    const float* hidden = (const float*)d_in[0];
    const float* mask   = (const float*)d_in[1];
    const float* phi    = (const float*)d_in[2];
    const float* Wq = (const float*)d_in[3];  const float* bq = (const float*)d_in[4];
    const float* Wk = (const float*)d_in[5];  const float* bk = (const float*)d_in[6];
    const float* Wv = (const float*)d_in[7];  const float* bv = (const float*)d_in[8];
    const float* Wo = (const float*)d_in[9];  const float* bo = (const float*)d_in[10];
    const float* gamma = (const float*)d_in[11];
    const float* beta  = (const float*)d_in[12];
    float* out = (float*)d_out;

    // workspace layout (88 MB total, with dead-buffer reuse)
    char* ws = (char*)d_ws;
    const size_t MB = 1024u * 1024u;
    float* q_f = (float*)(ws + 0 * MB);            // 16 MB (dead after rotary)
    float* k_f = (float*)(ws + 16 * MB);           // 16 MB (dead after rotary)
    float* v_f = (float*)(ws + 32 * MB);           // 16 MB
    unsigned short* hid_bf = (unsigned short*)(ws + 48 * MB);  // 8 MB
    unsigned short* Wq_bf  = (unsigned short*)(ws + 56 * MB);  // 2 MB
    unsigned short* Wk_bf  = (unsigned short*)(ws + 58 * MB);
    unsigned short* Wv_bf  = (unsigned short*)(ws + 60 * MB);
    unsigned short* Wo_bf  = (unsigned short*)(ws + 62 * MB);
    unsigned short* qr_bf  = (unsigned short*)(ws + 64 * MB);  // 8 MB
    unsigned short* kr_bf  = (unsigned short*)(ws + 72 * MB);  // 8 MB
    unsigned short* vt_bf  = (unsigned short*)(ws + 80 * MB);  // 8 MB
    float* preln           = (float*)(ws + 0 * MB);            // reuses q_f
    unsigned short* ctx_bf = (unsigned short*)(ws + 16 * MB);  // reuses k_f

    // 1) f32 -> bf16 converts
    {
        const int nHid4 = (N * D) / 4, nW4 = (D * D) / 4;
        dim3 blk(256);
        convert_f32_bf16_k<<<dim3((nHid4 + 255) / 256), blk, 0, stream>>>(
            (const v4f*)hidden, (v2u*)hid_bf, nHid4);
        convert_f32_bf16_k<<<dim3((nW4 + 255) / 256), blk, 0, stream>>>(
            (const v4f*)Wq, (v2u*)Wq_bf, nW4);
        convert_f32_bf16_k<<<dim3((nW4 + 255) / 256), blk, 0, stream>>>(
            (const v4f*)Wk, (v2u*)Wk_bf, nW4);
        convert_f32_bf16_k<<<dim3((nW4 + 255) / 256), blk, 0, stream>>>(
            (const v4f*)Wv, (v2u*)Wv_bf, nW4);
        convert_f32_bf16_k<<<dim3((nW4 + 255) / 256), blk, 0, stream>>>(
            (const v4f*)Wo, (v2u*)Wo_bf, nW4);
    }

    // 2) QKV projections (WMMA bf16 GEMMs)
    {
        dim3 blk(128), grd(D / 128, N / 128);   // (8, 32)
        gemm_bf16_k<<<grd, blk, 0, stream>>>(hid_bf, Wq_bf, bq, nullptr, q_f, N, D, D);
        gemm_bf16_k<<<grd, blk, 0, stream>>>(hid_bf, Wk_bf, bk, nullptr, k_f, N, D, D);
        gemm_bf16_k<<<grd, blk, 0, stream>>>(hid_bf, Wv_bf, bv, nullptr, v_f, N, D, D);
    }

    // 3) rotary + head-major bf16 packing (+ V transpose)
    {
        const int total = B * H * L * 32;
        rotary_k<<<dim3((total + 255) / 256), dim3(256), 0, stream>>>(
            q_f, k_f, v_f, phi, qr_bf, kr_bf, vt_bf, B, H, L);
    }

    // 4) attention (WMMA, online softmax, async-LDS staged K/V)
    {
        dim3 blk(128), grd(L / 64, B * H);      // (32, 32)
        attention_k<<<grd, blk, 0, stream>>>(qr_bf, kr_bf, vt_bf, mask, ctx_bf, B, H, L);
    }

    // 5) output projection + bias + residual (WMMA GEMM)
    {
        dim3 blk(128), grd(D / 128, N / 128);
        gemm_bf16_k<<<grd, blk, 0, stream>>>(ctx_bf, Wo_bf, bo, hidden, preln, N, D, D);
    }

    // 6) LayerNorm -> d_out
    layernorm_k<<<dim3(N), dim3(256), 0, stream>>>(preln, gamma, beta, out, D);
}
